// HetNet4Layer_5231270166888
// MI455X (gfx1250) — compile-verified, matrix operand order checked by reading, not run on hardware
//
#include <hip/hip_runtime.h>

// ---------------------------------------------------------------------------
// Heterogeneous 4-layer GAT for MI455X (gfx1250, wave32, WMMA).
// Dense per-head projections -> single [N,K]x[K,128] bf16 WMMA GEMM (fp32 acc).
// Edge softmax + scatter stay fp32 (bandwidth-bound: ~3.2 GB irregular traffic
// total -> ~140us floor at 23.3 TB/s; GEMMs are tiny, so WMMA just keeps them
// off the critical path).
// ---------------------------------------------------------------------------

#define NN   20000     // nodes per type (uav == task == 20000)
#define EE   200000    // edges per etype
#define NH   4         // heads
#define DOUT 32        // per-head output dim
#define COLS 128       // NH * DOUT

typedef __bf16 bf16_t;
typedef __attribute__((ext_vector_type(16))) __bf16 v16bf;
typedef __attribute__((ext_vector_type(8)))  __bf16 v8bf;
typedef __attribute__((ext_vector_type(8)))  float  v8f;

// ---- order-preserving float<->uint encoding for atomicMax on floats --------
__device__ __forceinline__ unsigned encf(float f) {
  unsigned u = __float_as_uint(f);
  return (u & 0x80000000u) ? ~u : (u | 0x80000000u);
}
__device__ __forceinline__ float decf(unsigned u) {
  return __uint_as_float((u & 0x80000000u) ? (u & 0x7fffffffu) : ~u);
}

// ---------------------------------------------------------------------------
// Generic fills / conversions
// ---------------------------------------------------------------------------
__global__ void k_fill_u32(unsigned* __restrict__ p, unsigned v, size_t n) {
  size_t t = (size_t)blockIdx.x * blockDim.x + threadIdx.x;
  if (t < n) p[t] = v;
}

__global__ void k_f32_to_bf16(const float* __restrict__ in,
                              bf16_t* __restrict__ out, size_t n) {
  size_t t = (size_t)blockIdx.x * blockDim.x + threadIdx.x;
  if (t < n) out[t] = (bf16_t)in[t];
}

// Convert param W [NH, K, DOUT] (row-major f32) into bf16, column-contiguous:
// Wt[c*K + k] = W[h, k, o] with c = h*DOUT + o. Each output column's K values
// are contiguous, matching the per-lane B-fragment load pattern.
__global__ void k_w_conv(const float* __restrict__ W,
                         bf16_t* __restrict__ Wt, int K) {
  int t = blockIdx.x * blockDim.x + threadIdx.x;
  if (t >= COLS * K) return;
  int c = t / K, k = t - c * K;
  int h = c >> 5, o = c & 31;
  Wt[(size_t)c * K + k] = (bf16_t)W[((size_t)h * K + k) * DOUT + o];
}

// ---------------------------------------------------------------------------
// WMMA GEMM: C[N,COLS] = A[N,K](bf16) x Bt(bf16, col-contiguous [COLS][K]) + bias
// Block = 128 threads = 4 waves; wave w owns column tile ct = blockIdx.y*4+w.
// A frag (16x32 bf16): lane L reads row (m0 + L%16), 16 contiguous K starting
// at k0 + (L/16)*16. B frag symmetric on columns. C/D: M = r + (L/16)*8,
// N = L%16 (ISA 7.12.2).
// ---------------------------------------------------------------------------
__global__ __launch_bounds__(128) void k_gemm_bf16_wmma(
    const bf16_t* __restrict__ A, const bf16_t* __restrict__ Bt,
    const float* __restrict__ bias, float* __restrict__ C, int N, int K) {
  const int lane = threadIdx.x & 31;
  const int wv   = threadIdx.x >> 5;
  const int l16  = lane & 15;
  const int kh   = lane >> 4;            // 0/1: which K half of the fragment
  const int m0   = blockIdx.x * 16;
  const int ct   = blockIdx.y * 4 + wv;  // 0..7 column tiles of 16

  int arow = m0 + l16; if (arow > N - 1) arow = N - 1;   // EXEC must stay full
  const int bcol = ct * 16 + l16;

  const bf16_t* ap = A  + (size_t)arow * K + kh * 16;
  const bf16_t* bp = Bt + (size_t)bcol * K + kh * 16;

  v8f acc = {};
#pragma unroll 2
  for (int k0 = 0; k0 < K; k0 += 32) {
    union { v16bf v; v8bf h[2]; } a, b;
    a.h[0] = *(const v8bf*)(ap + k0);
    a.h[1] = *(const v8bf*)(ap + k0 + 8);
    b.h[0] = *(const v8bf*)(bp + k0);
    b.h[1] = *(const v8bf*)(bp + k0 + 8);
    acc = __builtin_amdgcn_wmma_f32_16x16x32_bf16(
        false, a.v, false, b.v, (short)0, acc, false, false);
  }
  const float bv = bias ? bias[bcol] : 0.f;
#pragma unroll
  for (int r = 0; r < 8; ++r) {
    int m = m0 + kh * 8 + r;
    if (m < N) C[(size_t)m * COLS + bcol] = acc[r] + bv;
  }
}

// ---------------------------------------------------------------------------
// es[n,h] = sum_o z[n, h*32+o] * a[h*32+o]
// ---------------------------------------------------------------------------
__global__ void k_attn_score(const float* __restrict__ z,
                             const float* __restrict__ a,
                             float* __restrict__ out, int N) {
  int t = blockIdx.x * blockDim.x + threadIdx.x;
  if (t >= N * NH) return;
  int n = t >> 2, h = t & 3;
  const float4* zr = (const float4*)(z + (size_t)n * COLS + h * DOUT);
  const float4* ar = (const float4*)(a + h * DOUT);
  float s = 0.f;
#pragma unroll
  for (int q = 0; q < DOUT / 4; ++q) {
    float4 zv = zr[q], av = ar[q];
    s += zv.x * av.x + zv.y * av.y + zv.z * av.z + zv.w * av.w;
  }
  out[t] = s;
}

// ---------------------------------------------------------------------------
// Edge pass 1: e = leaky_relu(es[si]+ed[di]); segment-max into m (encoded).
// ---------------------------------------------------------------------------
__global__ void k_edge_max(const int* __restrict__ si, const int* __restrict__ di,
                           const float* __restrict__ es, const float* __restrict__ ed,
                           float* __restrict__ eedge, unsigned* __restrict__ m) {
  int t = blockIdx.x * blockDim.x + threadIdx.x;
  if (t >= EE * NH) return;
  int e = t >> 2, h = t & 3;
  int s = si[e], d = di[e];
  float v = es[s * NH + h] + ed[d * NH + h];
  v = v > 0.f ? v : 0.2f * v;
  eedge[t] = v;
  atomicMax(&m[d * NH + h], encf(v));
}

// Edge pass 2: w = exp(e - m[di]); segment-sum into denom.
__global__ void k_edge_expsum(const int* __restrict__ di,
                              const float* __restrict__ eedge,
                              const unsigned* __restrict__ m,
                              float* __restrict__ wedge,
                              float* __restrict__ denom) {
  int t = blockIdx.x * blockDim.x + threadIdx.x;
  if (t >= EE * NH) return;
  int e = t >> 2, h = t & 3;
  int d = di[e];
  float w = __expf(eedge[t] - decf(m[d * NH + h]));
  wedge[t] = w;
  atomicAdd(&denom[d * NH + h], w);
}

// Edge pass 3: acc[di] += z_s[si] * alpha  (float4-granular, one h per group).
__global__ void k_edge_scatter(const int* __restrict__ si, const int* __restrict__ di,
                               const float* __restrict__ z,
                               const float* __restrict__ wedge,
                               const float* __restrict__ denom,
                               float* __restrict__ acc) {
  int t = blockIdx.x * blockDim.x + threadIdx.x;
  if (t >= EE * (COLS / 4)) return;
  int e = t >> 5, q = t & 31;            // q selects 4 columns: [q*4, q*4+3]
  int h = q >> 3;                        // all 4 columns share head h
  int s = si[e], d = di[e];
  float alpha = wedge[e * NH + h] / (denom[d * NH + h] + 1e-9f);
  float4 zv = ((const float4*)(z + (size_t)s * COLS))[q];
  float* ap = acc + (size_t)d * COLS + q * 4;
  atomicAdd(ap + 0, zv.x * alpha);
  atomicAdd(ap + 1, zv.y * alpha);
  atomicAdd(ap + 2, zv.z * alpha);
  atomicAdd(ap + 3, zv.w * alpha);
}

// Merge 'cat': relu then bf16 for the next layer's GEMM input.
__global__ void k_merge_cat(const float* __restrict__ acc,
                            bf16_t* __restrict__ feats, size_t n) {
  size_t t = (size_t)blockIdx.x * blockDim.x + threadIdx.x;
  if (t >= n) return;
  float v = acc[t];
  feats[t] = (bf16_t)(v > 0.f ? v : 0.f);
}

// Merge 'avg' (final layer): mean over heads, no activation, fp32 out.
__global__ void k_merge_avg(const float* __restrict__ acc,
                            float* __restrict__ out, int N) {
  int t = blockIdx.x * blockDim.x + threadIdx.x;
  if (t >= N * DOUT) return;
  int n = t >> 5, o = t & 31;
  const float* r = acc + (size_t)n * COLS;
  out[t] = 0.25f * (r[o] + r[32 + o] + r[64 + o] + r[96 + o]);
}

// ---------------------------------------------------------------------------
// Host orchestration
// ---------------------------------------------------------------------------
extern "C" void kernel_launch(void* const* d_in, const int* in_sizes, int n_in,
                              void* d_out, int out_size, void* d_ws, size_t ws_size,
                              hipStream_t stream) {
  (void)in_sizes; (void)n_in; (void)out_size; (void)ws_size;

  const float* feat_in[2] = { (const float*)d_in[0], (const float*)d_in[1] };
  // etypes in CETYPES order: u2u, u2t, t2u, t2t  (0 = uav, 1 = task)
  const int eS[4] = { 0, 0, 1, 1 };
  const int eD[4] = { 0, 1, 0, 1 };
  const int* esrc[4]; const int* edst[4];
  for (int ei = 0; ei < 4; ++ei) {
    esrc[ei] = (const int*)d_in[2 + 2 * ei];
    edst[ei] = (const int*)d_in[3 + 2 * ei];
  }
  // params flattened in insertion order: layer -> etype -> {W_src,b_src,W_dst,a_src,a_dst}
  auto P = [&](int li, int ei, int j) -> const float* {
    return (const float*)d_in[10 + li * 20 + ei * 5 + j];
  };

  // Workspace carve-out (hipMalloc'd base is 256B aligned).
  char* w = (char*)d_ws;
  auto carve = [&](size_t bytes) -> char* {
    char* p = w; w += (bytes + 255) & ~(size_t)255; return p;
  };
  bf16_t* feats[2]; // bf16 node features, stride = current K
  feats[0] = (bf16_t*)carve((size_t)NN * COLS * 2);
  feats[1] = (bf16_t*)carve((size_t)NN * COLS * 2);
  float* z_s   = (float*)carve((size_t)NN * COLS * 4);
  float* z_d   = (float*)carve((size_t)NN * COLS * 4);
  float* acc[2];
  acc[0] = (float*)carve((size_t)NN * COLS * 4);
  acc[1] = (float*)carve((size_t)NN * COLS * 4);
  float*    es    = (float*)carve((size_t)NN * NH * 4);
  float*    edv   = (float*)carve((size_t)NN * NH * 4);
  unsigned* segm  = (unsigned*)carve((size_t)NN * NH * 4);
  float*    denom = (float*)carve((size_t)NN * NH * 4);
  float*    eedge = (float*)carve((size_t)EE * NH * 4);
  float*    wedge = (float*)carve((size_t)EE * NH * 4);
  bf16_t*   Wt_s  = (bf16_t*)carve((size_t)COLS * COLS * 2);
  bf16_t*   Wt_d  = (bf16_t*)carve((size_t)COLS * COLS * 2);

  auto NB = [](size_t n, int b) { return (unsigned)((n + b - 1) / b); };
  const int B = 256;

  // Layer 0 inputs: f32 [NN,64] -> bf16 (stride 64).
  k_f32_to_bf16<<<NB((size_t)NN * 64, B), B, 0, stream>>>(feat_in[0], feats[0], (size_t)NN * 64);
  k_f32_to_bf16<<<NB((size_t)NN * 64, B), B, 0, stream>>>(feat_in[1], feats[1], (size_t)NN * 64);

  int K = 64;
  for (int li = 0; li < 4; ++li) {
    // zero per-dst-type accumulators
    k_fill_u32<<<NB((size_t)NN * COLS, B), B, 0, stream>>>((unsigned*)acc[0], 0u, (size_t)NN * COLS);
    k_fill_u32<<<NB((size_t)NN * COLS, B), B, 0, stream>>>((unsigned*)acc[1], 0u, (size_t)NN * COLS);

    for (int ei = 0; ei < 4; ++ei) {
      const int s = eS[ei], d = eD[ei];
      // weights -> bf16 fragment layout
      k_w_conv<<<NB((size_t)COLS * K, B), B, 0, stream>>>(P(li, ei, 0), Wt_s, K);
      k_w_conv<<<NB((size_t)COLS * K, B), B, 0, stream>>>(P(li, ei, 2), Wt_d, K);
      // projections (WMMA): z_s = feats[s]*W_src + b_src ; z_d = feats[d]*W_dst
      dim3 gg((NN + 15) / 16, 2);
      k_gemm_bf16_wmma<<<gg, 128, 0, stream>>>(feats[s], Wt_s, P(li, ei, 1), z_s, NN, K);
      k_gemm_bf16_wmma<<<gg, 128, 0, stream>>>(feats[d], Wt_d, nullptr,      z_d, NN, K);
      // attention logits per node
      k_attn_score<<<NB((size_t)NN * NH, B), B, 0, stream>>>(z_s, P(li, ei, 3), es, NN);
      k_attn_score<<<NB((size_t)NN * NH, B), B, 0, stream>>>(z_d, P(li, ei, 4), edv, NN);
      // segment softmax state: m = 0u, below encf(-inf) ; denom = 0
      k_fill_u32<<<NB((size_t)NN * NH, B), B, 0, stream>>>(segm, 0u, (size_t)NN * NH);
      k_fill_u32<<<NB((size_t)NN * NH, B), B, 0, stream>>>((unsigned*)denom, 0u, (size_t)NN * NH);
      // edge passes
      k_edge_max<<<NB((size_t)EE * NH, B), B, 0, stream>>>(esrc[ei], edst[ei], es, edv, eedge, segm);
      k_edge_expsum<<<NB((size_t)EE * NH, B), B, 0, stream>>>(edst[ei], eedge, segm, wedge, denom);
      k_edge_scatter<<<NB((size_t)EE * (COLS / 4), B), B, 0, stream>>>(
          esrc[ei], edst[ei], z_s, wedge, denom, acc[d]);
    }

    if (li < 3) {
      k_merge_cat<<<NB((size_t)NN * COLS, B), B, 0, stream>>>(acc[0], feats[0], (size_t)NN * COLS);
      k_merge_cat<<<NB((size_t)NN * COLS, B), B, 0, stream>>>(acc[1], feats[1], (size_t)NN * COLS);
      K = COLS;
    } else {
      float* out = (float*)d_out;
      k_merge_avg<<<NB((size_t)NN * DOUT, B), B, 0, stream>>>(acc[0], out, NN);
      k_merge_avg<<<NB((size_t)NN * DOUT, B), B, 0, stream>>>(acc[1], out + (size_t)NN * DOUT, NN);
    }
  }
}
